// GNNDSE_78666620994254
// MI455X (gfx1250) — matrix operand order repaired and checked by reading, not compiled, not run on hardware
//
#include <hip/hip_runtime.h>
#include <hip/hip_bf16.h>
#include <math.h>

#define N_NODES  100000
#define N_EDGES  1600000
#define N_GRAPHS 64
#define DD       64
#define IN_CH    128

typedef __attribute__((ext_vector_type(16))) _Float16 v16h;
typedef __attribute__((ext_vector_type(8)))  float    v8f;

// ---------- helpers ----------
__device__ __forceinline__ unsigned f2key(float f) {
    unsigned b = __float_as_uint(f);
    return (b & 0x80000000u) ? ~b : (b | 0x80000000u);
}
__device__ __forceinline__ float key2f(unsigned u) {
    unsigned b = (u & 0x80000000u) ? (u ^ 0x80000000u) : ~u;
    return __uint_as_float(b);
}
__device__ __forceinline__ float eluf(float x) { return x > 0.f ? x : (expf(x) - 1.f); }

// ---------- WMMA GEMM: C[M x 64] = A[M x K] @ W[K x 64] + bias, optional relu ----------
// Requirements (all call sites satisfy): M % 16 == 0, K % 32 == 0, K <= 128,
// A and C 16-byte aligned.
// One wave computes a 16x64 tile (4 accumulators); 4 waves per block -> 64 rows/block.
// W is staged into LDS transposed + converted to f16 so each lane's B fragment is
// two runs of 8 contiguous halfs (ds_load_b128 x2); A fragments are 4x global_load_b128.
#define GEMM_WAVES 4
#define SW_STRIDE  132   // padded (vs 128) to spread LDS banks

__global__ void gemm64_wmma_k(const float* __restrict__ A, int K,
                              const float* __restrict__ W,
                              const float* __restrict__ bias,
                              float* __restrict__ C, int M, int relu)
{
    __shared__ _Float16 sW[64 * SW_STRIDE];   // sW[col * SW_STRIDE + k] = W[k][col]
    for (int i = threadIdx.x; i < K * 64; i += blockDim.x) {
        int k = i >> 6, col = i & 63;
        sW[col * SW_STRIDE + k] = (_Float16)W[i];
    }
    __syncthreads();

    int wave    = threadIdx.x >> 5;
    int lane    = threadIdx.x & 31;
    int tileM   = blockIdx.x * GEMM_WAVES + wave;
    int rowBase = tileM * 16;
    if (rowBase >= M) return;
    int r16 = lane & 15;
    int g   = lane >> 4;
    int row = rowBase + r16;

    v8f acc[4];
#pragma unroll
    for (int t = 0; t < 4; ++t) {
        float bv = bias[t * 16 + r16];
#pragma unroll
        for (int i = 0; i < 8; ++i) acc[t][i] = bv;
    }

    const float* ap = A + (size_t)row * K;
    for (int kc = 0; kc < K; kc += 32) {
        // A fragment: lane needs K = [kc+8g, kc+8g+8) and [kc+16+8g, kc+16+8g+8)
        float4 a01 = *(const float4*)(ap + kc + 8 * g);
        float4 a23 = *(const float4*)(ap + kc + 8 * g + 4);
        float4 a45 = *(const float4*)(ap + kc + 16 + 8 * g);
        float4 a67 = *(const float4*)(ap + kc + 16 + 8 * g + 4);
        v16h a;
        a[0]  = (_Float16)a01.x; a[1]  = (_Float16)a01.y;
        a[2]  = (_Float16)a01.z; a[3]  = (_Float16)a01.w;
        a[4]  = (_Float16)a23.x; a[5]  = (_Float16)a23.y;
        a[6]  = (_Float16)a23.z; a[7]  = (_Float16)a23.w;
        a[8]  = (_Float16)a45.x; a[9]  = (_Float16)a45.y;
        a[10] = (_Float16)a45.z; a[11] = (_Float16)a45.w;
        a[12] = (_Float16)a67.x; a[13] = (_Float16)a67.y;
        a[14] = (_Float16)a67.z; a[15] = (_Float16)a67.w;

#pragma unroll
        for (int t = 0; t < 4; ++t) {
            int col = t * 16 + r16;
            const _Float16* wp = &sW[col * SW_STRIDE + kc + 8 * g];
            v16h b;
#pragma unroll
            for (int i = 0; i < 16; ++i) b[i] = wp[i < 8 ? i : i + 8];
            acc[t] = __builtin_amdgcn_wmma_f32_16x16x32_f16(false, a, false, b,
                                                            (short)0, acc[t], false, false);
        }
    }

#pragma unroll
    for (int t = 0; t < 4; ++t) {
        int col = t * 16 + r16;
#pragma unroll
        for (int r = 0; r < 8; ++r) {
            int orow = rowBase + r + 8 * g;   // C/D layout: lane<16 -> M=r, lane>=16 -> M=r+8
            float vout = acc[t][r];
            if (relu) vout = fmaxf(vout, 0.f);
            C[(size_t)orow * 64 + col] = vout;
        }
    }
}

// ---------- per-layer segment scratch init ----------
__global__ void init_seg_k(unsigned* __restrict__ key, float* __restrict__ sum)
{
    int n = blockIdx.x * blockDim.x + threadIdx.x;
    if (n < N_NODES) { key[n] = 0u; sum[n] = 0.f; }
}

__global__ void zero32_k(unsigned* __restrict__ p, int n)
{
    int i = blockIdx.x * blockDim.x + threadIdx.x;
    if (i < n) p[i] = 0u;
}

// ---------- edge pass 1: attention logits + segment max ----------
__global__ void edge_alpha_k(const int* __restrict__ src, const int* __restrict__ dst,
                             const float* __restrict__ ea, const float* __restrict__ We,
                             const float* __restrict__ q,  const float* __restrict__ kmat,
                             float* __restrict__ alpha, unsigned* __restrict__ segkey)
{
    __shared__ float sWe[7 * 64];
    for (int i = threadIdx.x; i < 7 * 64; i += blockDim.x) sWe[i] = We[i];
    __syncthreads();
    int e = blockIdx.x * blockDim.x + threadIdx.x;
    if (e >= N_EDGES) return;
    {
        int ep = e + 16384; if (ep >= N_EDGES) ep = N_EDGES - 1;
        __builtin_prefetch(ea + (size_t)ep * 7, 0, 1);   // global_prefetch_b8
    }
    int s = src[e], d = dst[e];
    const float* eap = ea + (size_t)e * 7;
    float a0 = eap[0], a1 = eap[1], a2 = eap[2], a3 = eap[3], a4 = eap[4], a5 = eap[5], a6 = eap[6];
    const float* qp = q    + (size_t)d * 64;
    const float* kp = kmat + (size_t)s * 64;
    float acc = 0.f;
#pragma unroll 8
    for (int c = 0; c < 64; ++c) {
        float ed = a0 * sWe[c]       + a1 * sWe[64 + c]  + a2 * sWe[128 + c] +
                   a3 * sWe[192 + c] + a4 * sWe[256 + c] + a5 * sWe[320 + c] +
                   a6 * sWe[384 + c];
        acc += qp[c] * (kp[c] + ed);
    }
    float lg = acc * 0.125f;  // / sqrt(64)
    alpha[e] = lg;
    atomicMax(&segkey[d], f2key(lg));
}

// ---------- edge pass 2: exp(logit - max) + segment sum ----------
__global__ void edge_exp_k(const int* __restrict__ dst, const unsigned* __restrict__ segkey,
                           float* __restrict__ alpha, float* __restrict__ segsum)
{
    int e = blockIdx.x * blockDim.x + threadIdx.x;
    if (e >= N_EDGES) return;
    int d = dst[e];
    float m = key2f(segkey[d]);
    float w = expf(alpha[e] - m);
    alpha[e] = w;
    atomicAdd(&segsum[d], w);
}

// ---------- edge pass 3: weighted message scatter-add ----------
__global__ void edge_msg_k(const int* __restrict__ src, const int* __restrict__ dst,
                           const float* __restrict__ ea, const float* __restrict__ We,
                           const float* __restrict__ vmat, const float* __restrict__ alpha,
                           const float* __restrict__ segsum, float* __restrict__ msg)
{
    __shared__ float sWe[7 * 64];
    for (int i = threadIdx.x; i < 7 * 64; i += blockDim.x) sWe[i] = We[i];
    __syncthreads();
    int e = blockIdx.x * blockDim.x + threadIdx.x;
    if (e >= N_EDGES) return;
    {
        int ep = e + 16384; if (ep >= N_EDGES) ep = N_EDGES - 1;
        __builtin_prefetch(ea + (size_t)ep * 7, 0, 1);
    }
    int s = src[e], d = dst[e];
    const float* eap = ea + (size_t)e * 7;
    float a0 = eap[0], a1 = eap[1], a2 = eap[2], a3 = eap[3], a4 = eap[4], a5 = eap[5], a6 = eap[6];
    float aw = alpha[e] / (segsum[d] + 1e-16f);
    const float* vp = vmat + (size_t)s * 64;
    float* mp = msg + (size_t)d * 64;
#pragma unroll 4
    for (int c = 0; c < 64; ++c) {
        float ed = a0 * sWe[c]       + a1 * sWe[64 + c]  + a2 * sWe[128 + c] +
                   a3 * sWe[192 + c] + a4 * sWe[256 + c] + a5 * sWe[320 + c] +
                   a6 * sWe[384 + c];
        atomicAdd(&mp[c], aw * (vp[c] + ed));
    }
}

// ---------- finalize: (msg + skip already in msg) -> optional elu -> h, jk = max(jk, h) ----------
__global__ void finalize_k(const float* __restrict__ msg, float* __restrict__ h,
                           float* __restrict__ jk, int apply_elu, int first)
{
    int i = blockIdx.x * blockDim.x + threadIdx.x;
    if (i >= N_NODES * 64) return;
    float x = msg[i];
    if (apply_elu) x = eluf(x);
    h[i]  = x;
    jk[i] = first ? x : fmaxf(jk[i], x);
}

// ---------- gate: logit[n] = U[n] . g2W + g2b ----------
__global__ void gate_logit_k(const float* __restrict__ U, const float* __restrict__ g2W,
                             const float* __restrict__ g2b, float* __restrict__ logit)
{
    __shared__ float sw[64];
    if (threadIdx.x < 64) sw[threadIdx.x] = g2W[threadIdx.x];
    __syncthreads();
    int n = blockIdx.x * blockDim.x + threadIdx.x;
    if (n >= N_NODES) return;
    const float* up = U + (size_t)n * 64;
    float s = g2b[0];
#pragma unroll 8
    for (int c = 0; c < 64; ++c) s += up[c] * sw[c];
    logit[n] = s;
}

__global__ void graph_max_k(const int* __restrict__ batch, const float* __restrict__ logit,
                            unsigned* __restrict__ gkey)
{
    int n = blockIdx.x * blockDim.x + threadIdx.x;
    if (n >= N_NODES) return;
    atomicMax(&gkey[batch[n]], f2key(logit[n]));
}

__global__ void graph_exp_k(const int* __restrict__ batch, const unsigned* __restrict__ gkey,
                            float* __restrict__ logit, float* __restrict__ gsum)
{
    int n = blockIdx.x * blockDim.x + threadIdx.x;
    if (n >= N_NODES) return;
    int b = batch[n];
    float w = expf(logit[n] - key2f(gkey[b]));
    logit[n] = w;
    atomicAdd(&gsum[b], w);
}

__global__ void pool_k(const int* __restrict__ batch, const float* __restrict__ w,
                       const float* __restrict__ gsum, const float* __restrict__ jk,
                       float* __restrict__ pooled)
{
    int n = blockIdx.x * blockDim.x + threadIdx.x;
    if (n >= N_NODES) return;
    int b = batch[n];
    float gate = w[n] / (gsum[b] + 1e-16f);
    const float* jp = jk + (size_t)n * 64;
    float* pp = pooled + (size_t)b * 64;
#pragma unroll 4
    for (int c = 0; c < 64; ++c) atomicAdd(&pp[c], gate * jp[c]);
}

// ---------- 6-expert MLP ensemble: pooled[G,64] -> preds[G,6] ----------
__global__ void mlp_k(const float* __restrict__ pooled,
                      const float* __restrict__ m0W, const float* __restrict__ m0b,
                      const float* __restrict__ m1W, const float* __restrict__ m1b,
                      const float* __restrict__ m2W, const float* __restrict__ m2b,
                      const float* __restrict__ m3W, const float* __restrict__ m3b,
                      float* __restrict__ preds)
{
    int t = threadIdx.x;
    if (t >= 6 * N_GRAPHS) return;
    int gidx = t & 63;
    int ex   = t >> 6;
    float in[64];
#pragma unroll
    for (int d = 0; d < 64; ++d) in[d] = pooled[gidx * 64 + d];
    float h0[32];
    for (int hh = 0; hh < 32; ++hh) {
        float s = m0b[ex * 32 + hh];
        for (int d = 0; d < 64; ++d) s += in[d] * m0W[(size_t)ex * 2048 + d * 32 + hh];
        h0[hh] = eluf(s);
    }
    float h1[16];
    for (int kk = 0; kk < 16; ++kk) {
        float s = m1b[ex * 16 + kk];
        for (int hh = 0; hh < 32; ++hh) s += h0[hh] * m1W[(size_t)ex * 512 + hh * 16 + kk];
        h1[kk] = eluf(s);
    }
    float h2[8];
    for (int j = 0; j < 8; ++j) {
        float s = m2b[ex * 8 + j];
        for (int kk = 0; kk < 16; ++kk) s += h1[kk] * m2W[(size_t)ex * 128 + kk * 8 + j];
        h2[j] = eluf(s);
    }
    float o = m3b[ex];
    for (int j = 0; j < 8; ++j) o += h2[j] * m3W[ex * 8 + j];
    preds[gidx * 6 + ex] = o;   // preds[:, :, 0].T -> [G, 6]
}

__global__ void loss_k(const float* __restrict__ preds, const float* __restrict__ y,
                       float* __restrict__ out_total)
{
    if (blockIdx.x != 0 || threadIdx.x != 0) return;
    float tot = 0.f;
    for (int j = 0; j < 6; ++j) {
        float ss = 0.f;
        for (int g = 0; g < N_GRAPHS; ++g) {
            float d = preds[g * 6 + j] - y[g * 6 + j];
            ss += d * d;
        }
        tot += sqrtf(ss / (float)N_GRAPHS);
    }
    out_total[0] = tot;
}

// ---------- orchestration ----------
static inline void launch_gemm(const float* A, int K, const float* W, const float* b,
                               float* C, int M, int relu, hipStream_t stream)
{
    int tiles = M / 16;                       // M % 16 == 0 at all call sites
    dim3 grid((tiles + GEMM_WAVES - 1) / GEMM_WAVES);
    gemm64_wmma_k<<<grid, 32 * GEMM_WAVES, 0, stream>>>(A, K, W, b, C, M, relu);
}

extern "C" void kernel_launch(void* const* d_in, const int* in_sizes, int n_in,
                              void* d_out, int out_size, void* d_ws, size_t ws_size,
                              hipStream_t stream)
{
    const float* x     = (const float*)d_in[0];
    const int*   eidx  = (const int*)  d_in[1];
    const float* eattr = (const float*)d_in[2];
    const int*   batch = (const int*)  d_in[3];
    const float* y     = (const float*)d_in[4];
    const float* Wq0 = (const float*)d_in[5];  const float* bq0 = (const float*)d_in[6];
    const float* Wk0 = (const float*)d_in[7];  const float* bk0 = (const float*)d_in[8];
    const float* Wv0 = (const float*)d_in[9];  const float* bv0 = (const float*)d_in[10];
    const float* We0 = (const float*)d_in[11];
    const float* Ws0 = (const float*)d_in[12]; const float* bs0 = (const float*)d_in[13];
    const float* Wq  = (const float*)d_in[14]; const float* bq  = (const float*)d_in[15];
    const float* Wk  = (const float*)d_in[16]; const float* bk  = (const float*)d_in[17];
    const float* Wv  = (const float*)d_in[18]; const float* bv  = (const float*)d_in[19];
    const float* We  = (const float*)d_in[20];
    const float* Ws  = (const float*)d_in[21]; const float* bs  = (const float*)d_in[22];
    const float* g1W = (const float*)d_in[23]; const float* g1b = (const float*)d_in[24];
    const float* g2W = (const float*)d_in[25]; const float* g2b = (const float*)d_in[26];
    const float* m0W = (const float*)d_in[27]; const float* m0b = (const float*)d_in[28];
    const float* m1W = (const float*)d_in[29]; const float* m1b = (const float*)d_in[30];
    const float* m2W = (const float*)d_in[31]; const float* m2b = (const float*)d_in[32];
    const float* m3W = (const float*)d_in[33]; const float* m3b = (const float*)d_in[34];

    const int* src = eidx;
    const int* dst = eidx + N_EDGES;

    // workspace carve (floats)
    size_t N64 = (size_t)N_NODES * 64;
    float*    q      = (float*)d_ws;
    float*    kbuf   = q + N64;
    float*    vbuf   = kbuf + N64;
    float*    msg    = vbuf + N64;              // skip written here, messages atomicAdd on top
    float*    h      = msg + N64;
    float*    jk     = h + N64;
    float*    alpha  = jk + N64;                // N_EDGES
    unsigned* segkey = (unsigned*)(alpha + N_EDGES);   // N_NODES
    float*    segsum = (float*)(segkey + N_NODES);     // N_NODES
    float*    logit  = segsum + N_NODES;               // N_NODES (reused as softmax weights)
    unsigned* gkey   = (unsigned*)(logit + N_NODES);   // 64
    float*    gsum   = (float*)(gkey + N_GRAPHS);      // 64
    float*    pooled = gsum + N_GRAPHS;                // 64*64

    dim3 eblk(256), egrd((N_EDGES + 255) / 256);
    dim3 nblk(256), ngrd((N_NODES + 255) / 256);
    dim3 fgrd((N_NODES * 64 + 255) / 256);

    for (int l = 0; l < 4; ++l) {
        const float* A  = (l == 0) ? x : h;
        int          K  = (l == 0) ? IN_CH : DD;
        int          i  = l - 1;
        const float* Wq_ = (l == 0) ? Wq0 : Wq + (size_t)i * 64 * 64;
        const float* bq_ = (l == 0) ? bq0 : bq + (size_t)i * 64;
        const float* Wk_ = (l == 0) ? Wk0 : Wk + (size_t)i * 64 * 64;
        const float* bk_ = (l == 0) ? bk0 : bk + (size_t)i * 64;
        const float* Wv_ = (l == 0) ? Wv0 : Wv + (size_t)i * 64 * 64;
        const float* bv_ = (l == 0) ? bv0 : bv + (size_t)i * 64;
        const float* We_ = (l == 0) ? We0 : We + (size_t)i * 7 * 64;
        const float* Ws_ = (l == 0) ? Ws0 : Ws + (size_t)i * 64 * 64;
        const float* bs_ = (l == 0) ? bs0 : bs + (size_t)i * 64;

        launch_gemm(A, K, Wq_, bq_, q,    N_NODES, 0, stream);
        launch_gemm(A, K, Wk_, bk_, kbuf, N_NODES, 0, stream);
        launch_gemm(A, K, Wv_, bv_, vbuf, N_NODES, 0, stream);
        launch_gemm(A, K, Ws_, bs_, msg,  N_NODES, 0, stream);  // skip baseline

        init_seg_k<<<ngrd, nblk, 0, stream>>>(segkey, segsum);
        edge_alpha_k<<<egrd, eblk, 0, stream>>>(src, dst, eattr, We_, q, kbuf, alpha, segkey);
        edge_exp_k<<<egrd, eblk, 0, stream>>>(dst, segkey, alpha, segsum);
        edge_msg_k<<<egrd, eblk, 0, stream>>>(src, dst, eattr, We_, vbuf, alpha, segsum, msg);
        finalize_k<<<fgrd, nblk, 0, stream>>>(msg, h, jk, (l < 3) ? 1 : 0, (l == 0) ? 1 : 0);
    }

    // gate pooling
    launch_gemm(jk, DD, g1W, g1b, q /*reuse as relu hidden*/, N_NODES, 1, stream);
    gate_logit_k<<<ngrd, nblk, 0, stream>>>(q, g2W, g2b, logit);
    zero32_k<<<dim3((N_GRAPHS * 2 + N_GRAPHS * 64 + 255) / 256), 256, 0, stream>>>(
        gkey, N_GRAPHS * 2 + N_GRAPHS * 64);   // gkey + gsum + pooled contiguous
    graph_max_k<<<ngrd, nblk, 0, stream>>>(batch, logit, gkey);
    graph_exp_k<<<ngrd, nblk, 0, stream>>>(batch, gkey, logit, gsum);
    pool_k<<<ngrd, nblk, 0, stream>>>(batch, logit, gsum, jk, pooled);

    // heads + loss
    float* preds = (float*)d_out;
    mlp_k<<<1, 384, 0, stream>>>(pooled, m0W, m0b, m1W, m1b, m2W, m2b, m3W, m3b, preds);
    loss_k<<<1, 32, 0, stream>>>(preds, y, preds + N_GRAPHS * 6);
}
